// MHSA_45406394253361
// MI455X (gfx1250) — matrix operand order repaired and checked by reading, not compiled, not run on hardware
//
#include <hip/hip_runtime.h>
#include <hip/hip_bf16.h>

typedef __attribute__((ext_vector_type(16))) _Float16 v16h;
typedef __attribute__((ext_vector_type(8)))  float    v8f;

#define HW 16384
#define CCH 64

union PkU { _Float16 h[2]; unsigned u; };
union BU  { unsigned u[8]; v16h h; };

// ---------------------------------------------------------------------------
// Kernel 1: KAN QKV. One block = (channel c, tile of 128 pixel-rows).
// Per wave: 16 rows, full 64-col output, K=64 -> 8 x v_wmma_f32_16x16x32_f16.
// A operand (h1 = relu(x*W1+b1)) generated directly in registers in contiguous k order
// (k = 16*hl + s); B operand (W2, f16) pre-packed in LDS fragment order.
// Epilogue: +b2, relu -> h2 in LDS, then the tiny 64x3 W3 projection, scattered
// to the interleaved qkv buffer (f = c*3 + j).
// ---------------------------------------------------------------------------
__global__ __launch_bounds__(256) void kan_qkv_kernel(
    const float* __restrict__ x,  const float* __restrict__ W1,
    const float* __restrict__ b1, const float* __restrict__ W2,
    const float* __restrict__ b2, const float* __restrict__ W3,
    const float* __restrict__ b3, float* __restrict__ qkv)
{
    const int mtile = blockIdx.x;          // 0..255 (128 rows each)
    const int c     = blockIdx.y;          // 0..63
    const int tid   = threadIdx.x;
    const int lane  = tid & 31;
    const int wave  = tid >> 5;
    const int lrow  = lane & 15;
    const int hl    = lane >> 4;           // lane half

    __shared__ float W1s[64], b1s[64], b2s[64], W3s[64 * 3], b3s[3], xs[128];
    __shared__ unsigned Bpack[2][4][8][32];   // [k-block][N-tile][vgpr][lane]
    __shared__ float h2s[128][65];            // padded to dodge bank conflicts

    if (tid < 64) {
        W1s[tid] = W1[c * 64 + tid];
        b1s[tid] = b1[c * 64 + tid];
        b2s[tid] = b2[c * 64 + tid];
    }
    if (tid < 192) W3s[tid] = W3[c * 192 + tid];
    if (tid < 3)   b3s[tid] = b3[c * 3 + tid];
    if (tid < 128) {
        int n = mtile * 128 + tid;
        int b = n >> 14, p = n & (HW - 1);
        xs[tid] = x[((size_t)b * CCH + c) * HW + p];
    }
    // Pack W2[c] (f32 [K=64][N=64]) into f16 pairs in per-lane fragment order.
    for (int s = tid; s < 2048; s += 256) {
        int kk = s >> 10, t = (s >> 8) & 3, i = (s >> 5) & 7, l = s & 31;
        int col = t * 16 + (l & 15);
        int K   = kk * 32 + ((l >> 4) << 4) + 2 * i;   // B: lanes0-15 K=0..15, lanes16-31 K=16..31
        float w0 = W2[((size_t)c * 64 + K    ) * 64 + col];
        float w1 = W2[((size_t)c * 64 + K + 1) * 64 + col];
        PkU p; p.h[0] = (_Float16)w0; p.h[1] = (_Float16)w1;
        Bpack[kk][t][i][l] = p.u;
    }
    __syncthreads();

    // A fragments: h1 row = wave*16 + lrow; contiguous K placement (k = 16*hl + s).
    float xv = xs[wave * 16 + lrow];
    v16h A[2];
#pragma unroll
    for (int kk = 0; kk < 2; ++kk) {
#pragma unroll
        for (int i = 0; i < 8; ++i) {
            int K = kk * 32 + hl * 16 + 2 * i;
            float a0 = fmaxf(xv * W1s[K]     + b1s[K],     0.f);
            float a1 = fmaxf(xv * W1s[K + 1] + b1s[K + 1], 0.f);
            A[kk][2 * i]     = (_Float16)a0;
            A[kk][2 * i + 1] = (_Float16)a1;
        }
    }

    v8f acc[4];
#pragma unroll
    for (int t = 0; t < 4; ++t)
#pragma unroll
        for (int i = 0; i < 8; ++i) acc[t][i] = 0.f;

#pragma unroll
    for (int t = 0; t < 4; ++t) {
#pragma unroll
        for (int kk = 0; kk < 2; ++kk) {
            BU bb;
#pragma unroll
            for (int i = 0; i < 8; ++i) bb.u[i] = Bpack[kk][t][i][lane];
            acc[t] = __builtin_amdgcn_wmma_f32_16x16x32_f16(
                false, A[kk], false, bb.h, (short)0, acc[t], false, false);
        }
    }

    // h2 = relu(acc + b2) -> LDS (D layout: M = v + 8*hl, N = lrow)
#pragma unroll
    for (int t = 0; t < 4; ++t) {
        int col = t * 16 + lrow;
        float bb2 = b2s[col];
#pragma unroll
        for (int v = 0; v < 8; ++v) {
            int row = wave * 16 + v + 8 * hl;
            h2s[row][col] = fmaxf(acc[t][v] + bb2, 0.f);
        }
    }
    __syncthreads();

    // o[row][j] = h2[row,:] . W3[:,j] + b3[j]; scatter f = c*3 + j
    for (int e = tid; e < 384; e += 256) {
        int row = e / 3, j = e - row * 3;
        float s = b3s[j];
#pragma unroll 8
        for (int k = 0; k < 64; ++k) s += h2s[row][k] * W3s[k * 3 + j];
        int n = mtile * 128 + row;
        int b = n >> 14, p = n & (HW - 1);
        int f = c * 3 + j;                         // 0..63 q, 64..127 k, 128..191 v
        qkv[((size_t)b * 192 + f) * HW + p] = s;
    }
}

// ---------------------------------------------------------------------------
// Kernel 2: per (b, head) block. One pass over n computes the 8x8 Gram matrix
// of q,k rows AND their sums-of-squares (fused L2 norms). Fixed-order wave +
// LDS reductions => deterministic (no float atomics). Then softmax -> attn.
// ---------------------------------------------------------------------------
__global__ __launch_bounds__(256) void attn_kernel(
    const float* __restrict__ qkv, const float* __restrict__ temperature,
    float* __restrict__ attn)
{
    const int b = blockIdx.x >> 3, h = blockIdx.x & 7;
    const int tid = threadIdx.x, lane = tid & 31, wave = tid >> 5;

    float G[8][8], qss[8], kss[8];
#pragma unroll
    for (int i = 0; i < 8; ++i) {
        qss[i] = 0.f; kss[i] = 0.f;
#pragma unroll
        for (int j = 0; j < 8; ++j) G[i][j] = 0.f;
    }
    const float* qb = qkv + ((size_t)b * 192 +      h * 8) * HW;
    const float* kb = qkv + ((size_t)b * 192 + 64 + h * 8) * HW;

    for (int n = tid; n < HW; n += 256) {
        float qv[8], kv[8];
#pragma unroll
        for (int i = 0; i < 8; ++i) { qv[i] = qb[i * HW + n]; kv[i] = kb[i * HW + n]; }
#pragma unroll
        for (int i = 0; i < 8; ++i) { qss[i] += qv[i] * qv[i]; kss[i] += kv[i] * kv[i]; }
#pragma unroll
        for (int i = 0; i < 8; ++i)
#pragma unroll
            for (int j = 0; j < 8; ++j) G[i][j] += qv[i] * kv[j];
    }

    __shared__ float red[80][8];
    float vals[80];
#pragma unroll
    for (int i = 0; i < 8; ++i)
#pragma unroll
        for (int j = 0; j < 8; ++j) vals[i * 8 + j] = G[i][j];
#pragma unroll
    for (int i = 0; i < 8; ++i) { vals[64 + i] = qss[i]; vals[72 + i] = kss[i]; }
#pragma unroll
    for (int idx = 0; idx < 80; ++idx) {
        float v = vals[idx];
        v += __shfl_xor(v, 16); v += __shfl_xor(v, 8);
        v += __shfl_xor(v, 4);  v += __shfl_xor(v, 2); v += __shfl_xor(v, 1);
        if (lane == 0) red[idx][wave] = v;
    }
    __syncthreads();
    __shared__ float fin[80];
    if (tid < 80) {
        float s = 0.f;
#pragma unroll
        for (int w = 0; w < 8; ++w) s += red[tid][w];
        fin[tid] = s;
    }
    __syncthreads();
    if (tid < 8) {
        int i = tid;
        float qn = fmaxf(sqrtf(fin[64 + i]), 1e-12f);
        float temp = temperature[h];
        float lg[8], m = -1e30f;
#pragma unroll
        for (int j = 0; j < 8; ++j) {
            float kn = fmaxf(sqrtf(fin[72 + j]), 1e-12f);
            lg[j] = fin[i * 8 + j] / (qn * kn) * temp;
            m = fmaxf(m, lg[j]);
        }
        float s = 0.f;
#pragma unroll
        for (int j = 0; j < 8; ++j) { lg[j] = __expf(lg[j] - m); s += lg[j]; }
        float inv = 1.f / s;
#pragma unroll
        for (int j = 0; j < 8; ++j)
            attn[(((size_t)b * 8 + h) * 8 + i) * 8 + j] = lg[j] * inv;
    }
}

// ---------------------------------------------------------------------------
// Kernel 2b: M_b = Wout @ A_b (A block-diagonal per head) -> [B][64][64]
// ---------------------------------------------------------------------------
__global__ __launch_bounds__(256) void mproj_kernel(
    const float* __restrict__ attn, const float* __restrict__ Wout,
    float* __restrict__ M)
{
    int b = blockIdx.x, tid = threadIdx.x;
    for (int e = tid; e < 4096; e += 256) {
        int o = e >> 6, d = e & 63;
        int hd = d >> 3, dl = d & 7;
        const float* A = attn + ((size_t)b * 8 + hd) * 64;   // A[i*8+dl]
        float s = 0.f;
#pragma unroll
        for (int i = 0; i < 8; ++i) s += Wout[o * 64 + hd * 8 + i] * A[i * 8 + dl];
        M[((size_t)b * 64 + o) * 64 + d] = s;
    }
}

// ---------------------------------------------------------------------------
// Kernel 3: final = M_b @ V_b + bout  -> d_out [b][c][hw] directly.
// Block = (b, 32-col n-tile); 8 waves = 4 o-tiles x 2 n-subtiles; K=64 -> 2 WMMA.
// ---------------------------------------------------------------------------
__global__ __launch_bounds__(256) void out_gemm_kernel(
    const float* __restrict__ qkv, const float* __restrict__ M,
    const float* __restrict__ bout, float* __restrict__ out)
{
    const int b = blockIdx.y;
    const int nbase = blockIdx.x * 32;
    const int tid = threadIdx.x, lane = tid & 31, wave = tid >> 5;
    const int lrow = lane & 15, hl = lane >> 4;

    __shared__ float Vs[64][33];
    __shared__ float Ms[64][64];
    __shared__ float bouts[64];

    if (tid < 64) bouts[tid] = bout[tid];
    for (int e = tid; e < 2048; e += 256) {
        int k = e >> 5, n = e & 31;
        Vs[k][n] = qkv[((size_t)b * 192 + 128 + k) * HW + nbase + n];
    }
    for (int e = tid; e < 4096; e += 256)
        Ms[e >> 6][e & 63] = M[(size_t)b * 4096 + e];
    __syncthreads();

    const int otile = wave & 3, nsub = wave >> 2;
    v8f acc;
#pragma unroll
    for (int i = 0; i < 8; ++i) acc[i] = 0.f;

#pragma unroll
    for (int kk = 0; kk < 2; ++kk) {
        v16h Af, Bf;
#pragma unroll
        for (int i = 0; i < 8; ++i) {
            int K = kk * 32 + hl * 16 + 2 * i;
            Af[2 * i]     = (_Float16)Ms[otile * 16 + lrow][K];
            Af[2 * i + 1] = (_Float16)Ms[otile * 16 + lrow][K + 1];
            int Kb = kk * 32 + hl * 16 + 2 * i;
            Bf[2 * i]     = (_Float16)Vs[Kb][nsub * 16 + lrow];
            Bf[2 * i + 1] = (_Float16)Vs[Kb + 1][nsub * 16 + lrow];
        }
        acc = __builtin_amdgcn_wmma_f32_16x16x32_f16(
            false, Af, false, Bf, (short)0, acc, false, false);
    }
#pragma unroll
    for (int v = 0; v < 8; ++v) {
        int o = otile * 16 + v + 8 * hl;
        int n = nbase + nsub * 16 + lrow;
        out[((size_t)b * CCH + o) * HW + n] = acc[v] + bouts[o];
    }
}

// ---------------------------------------------------------------------------
extern "C" void kernel_launch(void* const* d_in, const int* in_sizes, int n_in,
                              void* d_out, int out_size, void* d_ws, size_t ws_size,
                              hipStream_t stream)
{
    const float* x    = (const float*)d_in[0];
    const float* W1   = (const float*)d_in[1];
    const float* b1   = (const float*)d_in[2];
    const float* W2   = (const float*)d_in[3];
    const float* b2   = (const float*)d_in[4];
    const float* W3   = (const float*)d_in[5];
    const float* b3   = (const float*)d_in[6];
    const float* temp = (const float*)d_in[7];
    const float* Wout = (const float*)d_in[8];
    const float* bout = (const float*)d_in[9];

    float* qkv  = (float*)d_ws;                         // 2*192*16384 f32
    float* attn = qkv + (size_t)2 * 192 * HW;           // 2*8*8*8 f32
    float* M    = attn + 1024;                          // 2*64*64 f32

    dim3 g1(256, 64);
    kan_qkv_kernel<<<g1, 256, 0, stream>>>(x, W1, b1, W2, b2, W3, b3, qkv);
    attn_kernel<<<16, 256, 0, stream>>>(qkv, temp, attn);
    mproj_kernel<<<2, 256, 0, stream>>>(attn, Wout, M);
    dim3 g3(512, 2);
    out_gemm_kernel<<<g3, 256, 0, stream>>>(qkv, M, bout, (float*)d_out);
}
